// Hybrid_QuanvModel_44856638439553
// MI455X (gfx1250) — matrix operand order repaired and checked by reading, not compiled, not run on hardware
//
#include <hip/hip_runtime.h>
#include <hip/hip_bf16.h>

// ---------------------------------------------------------------------------
// MI455X (gfx1250) implementation notes
//  * wave32, WMMA 16x16x32 f16->f32 (codegen-confirmed builtin)
//  * Whole model ~5.6 GFLOP, scratch ~60MB -> L2 (192MB) resident; implicit
//    im2col GEMM for convs, async Global->LDS staged GEMM for FC layers.
//  * quanv: psi0[k] = p[k]*(-i)^popcount(k) with p real  =>  psi = V' p where
//    V' = U^3 * diag(phase). Two real 16x16 GEMMs per 16-patch tile == 2 WMMA.
//  * FC B-operand staged with GLOBAL_LOAD_ASYNC_TO_LDS_B128 (ASYNCcnt),
//    ordered with s_wait_dscnt / s_wait_asynccnt per ISA ch.5/ch.10.
// ---------------------------------------------------------------------------

typedef __attribute__((ext_vector_type(16))) _Float16 v16h;
typedef __attribute__((ext_vector_type(8)))  float    v8f;

__device__ __forceinline__ float2 cmulf(float2 a, float2 b) {
  return make_float2(a.x * b.x - a.y * b.y, a.x * b.y + a.y * b.x);
}
__device__ __forceinline__ float2 caddf(float2 a, float2 b) {
  return make_float2(a.x + b.x, a.y + b.y);
}

// ---------------------------------------------------------------------------
// conv1 (1ch, 8x8, stride 2) + tanh :  (256,1,186,186) -> (256,90,90)
// ---------------------------------------------------------------------------
__global__ void conv1_tanh_kernel(const float* __restrict__ x,
                                  const float* __restrict__ w,
                                  const float* __restrict__ bias,
                                  float* __restrict__ c1) {
  int idx = blockIdx.x * blockDim.x + threadIdx.x;
  if (idx >= 256 * 90 * 90) return;
  int b = idx / 8100;
  int s = idx - b * 8100;
  int oy = s / 90, ox = s - (s / 90) * 90;
  const float* xb = x + (size_t)b * 186 * 186 + (size_t)(2 * oy) * 186 + 2 * ox;
  float acc = bias[0];
#pragma unroll
  for (int r = 0; r < 8; ++r)
#pragma unroll
    for (int q = 0; q < 8; ++q)
      acc += xb[r * 186 + q] * w[r * 8 + q];
  c1[idx] = tanhf(acc);
}

// ---------------------------------------------------------------------------
// maxpool 4x4 stride 2 + * pi/2 :  (256,90,90) -> (256,44,44)
// ---------------------------------------------------------------------------
__global__ void pool1_scale_kernel(const float* __restrict__ c1,
                                   float* __restrict__ h0) {
  int idx = blockIdx.x * blockDim.x + threadIdx.x;
  if (idx >= 256 * 44 * 44) return;
  int b = idx / 1936;
  int s = idx - b * 1936;
  int y = s / 44, xx = s - y * 44;
  const float* cb = c1 + (size_t)b * 8100 + (size_t)(2 * y) * 90 + 2 * xx;
  float m = -3.4e38f;
#pragma unroll
  for (int r = 0; r < 4; ++r)
#pragma unroll
    for (int q = 0; q < 4; ++q)
      m = fmaxf(m, cb[r * 90 + q]);
  h0[idx] = m * 1.5707963267948966f;
}

// ---------------------------------------------------------------------------
// Build circuit matrix: U = (L3 L2 L1), U3 = U^3, fold (-i)^popcount phase,
// emit Vre/Vim as f16 [16][32] (K zero-padded 16->32), ready as WMMA A-frags.
// One block of 256 threads == one thread per 16x16 matrix element.
// ---------------------------------------------------------------------------
__global__ void unitary_kernel(const float* __restrict__ qw,
                               _Float16* __restrict__ Vre,
                               _Float16* __restrict__ Vim) {
  __shared__ float2 rot[4][2][2];
  __shared__ float2 Um[256], Lm[256], Tm[256];
  int tid = threadIdx.x;
  int r = tid >> 4, c = tid & 15;
  Um[tid] = make_float2(r == c ? 1.f : 0.f, 0.f);
  __syncthreads();

  for (int l = 0; l < 3; ++l) {
    if (tid < 4) {  // one thread per wire builds its Rot(phi,theta,omega)
      int q = tid;
      float phi = qw[(l * 4 + q) * 3 + 0];
      float th  = qw[(l * 4 + q) * 3 + 1];
      float om  = qw[(l * 4 + q) * 3 + 2];
      float cs = cosf(th * 0.5f), sn = sinf(th * 0.5f);
      float ap = (phi + om) * 0.5f, am = (phi - om) * 0.5f;
      rot[q][0][0] = make_float2(cosf(ap) * cs, -sinf(ap) * cs);   //  e^{-i(p+o)/2} c
      rot[q][0][1] = make_float2(-cosf(am) * sn, -sinf(am) * sn);  // -e^{+i(p-o)/2} s
      rot[q][1][0] = make_float2(cosf(am) * sn, -sinf(am) * sn);   //  e^{-i(p-o)/2} s
      rot[q][1][1] = make_float2(cosf(ap) * cs,  sinf(ap) * cs);   //  e^{+i(p+o)/2} c
    }
    __syncthreads();
    // L = kron(rot0, rot1, rot2, rot3)  (wire 0 = MSB)
    float2 v = make_float2(1.f, 0.f);
#pragma unroll
    for (int q = 0; q < 4; ++q) {
      int br = (r >> (3 - q)) & 1, bc = (c >> (3 - q)) & 1;
      v = cmulf(v, rot[q][br][bc]);
    }
    Lm[tid] = v;
    __syncthreads();
    // CNOTs: permutation matrices (involutive) => row permutation of L
    int range = l + 1;
    for (int i = 0; i < 4; ++i) {
      int ctrl = i, tgt = (i + range) & 3;
      int cb = (r >> (3 - ctrl)) & 1;
      int pr = cb ? (r ^ (1 << (3 - tgt))) : r;
      float2 t = Lm[pr * 16 + c];
      __syncthreads();
      Lm[tid] = t;
      __syncthreads();
    }
    // U = L @ U
    float2 s = make_float2(0.f, 0.f);
    for (int k = 0; k < 16; ++k) s = caddf(s, cmulf(Lm[r * 16 + k], Um[k * 16 + c]));
    __syncthreads();
    Tm[tid] = s;
    __syncthreads();
    Um[tid] = Tm[tid];
    __syncthreads();
  }
  // U3 = U @ U @ U
  float2 s = make_float2(0.f, 0.f);
  for (int k = 0; k < 16; ++k) s = caddf(s, cmulf(Um[r * 16 + k], Um[k * 16 + c]));
  __syncthreads();
  Tm[tid] = s;
  __syncthreads();
  float2 f = make_float2(0.f, 0.f);
  for (int k = 0; k < 16; ++k) f = caddf(f, cmulf(Tm[r * 16 + k], Um[k * 16 + c]));
  // fold phase(k) = (-i)^popcount(k) into column k:  V' = U3 * diag(phase)
  int m4 = __popc(c) & 3;
  float re, im;
  if      (m4 == 0) { re =  f.x; im =  f.y; }
  else if (m4 == 1) { re =  f.y; im = -f.x; }
  else if (m4 == 2) { re = -f.x; im = -f.y; }
  else              { re = -f.y; im =  f.x; }
  Vre[r * 32 + c] = (_Float16)re;
  Vim[r * 32 + c] = (_Float16)im;
  Vre[r * 32 + 16 + c] = (_Float16)0.f;  // K padding 16..31
  Vim[r * 32 + 16 + c] = (_Float16)0.f;
}

// ---------------------------------------------------------------------------
// quanv: 16 patches per wave, 2 WMMAs (real & imag), probs, Z expvals, ReLU.
// (256,44,44) -> (256,4,22,22).  7744 tiles == 968 blocks x 8 waves exactly.
// ---------------------------------------------------------------------------
__global__ void quanv_kernel(const float* __restrict__ h0,
                             const _Float16* __restrict__ Vre,
                             const _Float16* __restrict__ Vim,
                             float* __restrict__ qout) {
  __shared__ _Float16 Pb[8][32 * 16];  // [wave][K=32(pad)][N=16]
  int wv = threadIdx.x >> 5, lane = threadIdx.x & 31;
  int tile = blockIdx.x * 8 + wv;
  int lo = lane & 15, hi = lane >> 4;
  int b = 0, oy = 0, ox = 0;
  if (lane < 16) {
    int pi = tile * 16 + lane;
    b = pi / 484;
    int s = pi - b * 484;
    oy = s / 22; ox = s - oy * 22;
    const float* base = h0 + (size_t)b * 44 * 44;
    float a0 = base[(2 * oy) * 44 + 2 * ox];
    float a1 = base[(2 * oy) * 44 + 2 * ox + 1];
    float a2 = base[(2 * oy + 1) * 44 + 2 * ox];
    float a3 = base[(2 * oy + 1) * 44 + 2 * ox + 1];
    float cw[4], sw[4];
    cw[0] = cosf(a0 * 0.5f); sw[0] = sinf(a0 * 0.5f);
    cw[1] = cosf(a1 * 0.5f); sw[1] = sinf(a1 * 0.5f);
    cw[2] = cosf(a2 * 0.5f); sw[2] = sinf(a2 * 0.5f);
    cw[3] = cosf(a3 * 0.5f); sw[3] = sinf(a3 * 0.5f);
#pragma unroll
    for (int k = 0; k < 16; ++k) {
      float p = 1.f;
#pragma unroll
      for (int w = 0; w < 4; ++w) p *= ((k >> (3 - w)) & 1) ? sw[w] : cw[w];
      Pb[wv][k * 16 + lane] = (_Float16)p;
      Pb[wv][(k + 16) * 16 + lane] = (_Float16)0.f;  // K pad
    }
  }
  __syncthreads();

  // A-frags (16x32 f16): lane holds row m=lo; slots v: K = v*2 + 16*hi
  v16h ar, ai, bb;
#pragma unroll
  for (int v = 0; v < 8; ++v) {
    int k0 = (v << 1) + (hi << 4);
    ar[2 * v]     = Vre[lo * 32 + k0];
    ar[2 * v + 1] = Vre[lo * 32 + k0 + 1];
    ai[2 * v]     = Vim[lo * 32 + k0];
    ai[2 * v + 1] = Vim[lo * 32 + k0 + 1];
  }
  // B-frag (32x16 f16): lane holds col n=lo; slot t: K = t + 16*hi
#pragma unroll
  for (int t = 0; t < 16; ++t) bb[t] = Pb[wv][(t + (hi << 4)) * 16 + lo];

  v8f z = {};
  v8f yre = __builtin_amdgcn_wmma_f32_16x16x32_f16(false, ar, false, bb, (short)0, z, false, false);
  v8f yim = __builtin_amdgcn_wmma_f32_16x16x32_f16(false, ai, false, bb, (short)0, z, false, false);

  // probs -> PauliZ expvals; lane has rows j = v + 8*hi for patch n = lo
  float e0 = 0.f, e1 = 0.f, e2 = 0.f, e3 = 0.f;
#pragma unroll
  for (int v = 0; v < 8; ++v) {
    int j = v + (hi << 3);
    float pr = yre[v] * yre[v] + yim[v] * yim[v];
    e0 += ((j >> 3) & 1) ? -pr : pr;
    e1 += ((j >> 2) & 1) ? -pr : pr;
    e2 += ((j >> 1) & 1) ? -pr : pr;
    e3 += (j & 1) ? -pr : pr;
  }
  e0 += __shfl_xor(e0, 16, 32);
  e1 += __shfl_xor(e1, 16, 32);
  e2 += __shfl_xor(e2, 16, 32);
  e3 += __shfl_xor(e3, 16, 32);
  if (lane < 16) {
    float* ob = qout + (size_t)b * 4 * 484 + oy * 22 + ox;
    ob[0 * 484] = fmaxf(e0, 0.f);
    ob[1 * 484] = fmaxf(e1, 0.f);
    ob[2 * 484] = fmaxf(e2, 0.f);
    ob[3 * 484] = fmaxf(e3, 0.f);
  }
}

// ---------------------------------------------------------------------------
// Generic implicit-GEMM WMMA kernel.  Y[m,n] = relu(sum_k W[m,k]*X(n,k) + b[m])
// One wave == one 16x16 output tile; K chunked by 32 (f16 operands, f32 acc).
// IS_CONV: X is (B,C,H,W), n=(b,oy,ox), k=(c,r,q) 4x4 kernel; scalar gather
//          (L2-resident), weight-row prefetch for next chunk.
// !IS_CONV: X row-major [N][K]; when K%32==0 the 16x32 f32 sub-tile is staged
//          to LDS with GLOBAL_LOAD_ASYNC_TO_LDS_B128 (ASYNCcnt), else scalar.
// Store: Y[bi*M*OHW + m*OHW + sp], OHW=OH*OW (==1 for FC).
// ---------------------------------------------------------------------------
template <bool IS_CONV>
__global__ void gemm_wmma_kernel(const float* __restrict__ Wt,
                                 const float* __restrict__ X,
                                 const float* __restrict__ bias,
                                 float* __restrict__ Y,
                                 int M_real, int K_real,
                                 int Mtiles, int totalTiles, int relu,
                                 int C, int H, int Wd, int OH, int OW, int stride) {
  __shared__ float Xs[8][16 * 32];  // per-wave FC staging tile (2KB/wave)
  int wv = threadIdx.x >> 5, lane = threadIdx.x & 31;
  int tile = blockIdx.x * (blockDim.x >> 5) + wv;
  if (tile >= totalTiles) return;
  int mt = tile % Mtiles, nt = tile / Mtiles;
  int lo = lane & 15, hi = lane >> 4;
  int m = mt * 16 + lo;  // A row this lane loads
  int n = nt * 16 + lo;  // B col this lane loads

  const float* xb;
  int OHW = IS_CONV ? OH * OW : 1;
  int bimg = n / OHW, sp = n - bimg * OHW;
  if (IS_CONV) {
    int oy = sp / OW, ox = sp - oy * OW;
    xb = X + (size_t)bimg * C * H * Wd + (size_t)(oy * stride) * Wd + (ox * stride);
  } else {
    xb = X + (size_t)n * K_real;
  }
  bool mok = (m < M_real);
  const float* wrow = Wt + (size_t)(mok ? m : 0) * K_real;
  bool asyncOK = (!IS_CONV) && ((K_real & 31) == 0);
  unsigned ldsbase = (unsigned)(size_t)(&Xs[wv][0]);

  v8f acc = {};
  int kChunks = (K_real + 31) >> 5;
  for (int kc = 0; kc < kChunks; ++kc) {
    int kbase = kc << 5;
    // prefetch next weight-row chunk (global_prefetch_b8)
    if (kbase + 32 < K_real) __builtin_prefetch(wrow + kbase + 32, 0, 0);

    v16h a, bb;
#pragma unroll
    for (int v = 0; v < 8; ++v) {
      int k0 = kbase + (v << 1) + (hi << 4);
      float f0 = (mok && k0 < K_real) ? wrow[k0] : 0.f;
      float f1 = (mok && (k0 + 1) < K_real) ? wrow[k0 + 1] : 0.f;
      a[2 * v] = (_Float16)f0;
      a[2 * v + 1] = (_Float16)f1;
    }

    if (asyncOK) {
      // make sure prior DS reads of Xs have completed before overwriting
      asm volatile("s_wait_dscnt 0x0" ::: "memory");
      // stage rows n0..n0+15, k kbase..kbase+31 (f32) : 128 x 16B segments
#pragma unroll
      for (int it = 0; it < 4; ++it) {
        int seg = it * 32 + lane;          // 0..127
        int row = seg >> 3, part = seg & 7;
        const float* src = X + (size_t)(nt * 16 + row) * K_real + kbase + part * 4;
        unsigned dst = ldsbase + (unsigned)seg * 16u;
        asm volatile("global_load_async_to_lds_b128 %0, %1, off"
                     :: "v"(dst), "v"(src) : "memory");
      }
      asm volatile("s_wait_asynccnt 0x0" ::: "memory");
#pragma unroll
      for (int t = 0; t < 16; ++t) {
        int k = t + (hi << 4);
        bb[t] = (_Float16)Xs[wv][lo * 32 + k];
      }
    } else {
#pragma unroll
      for (int t = 0; t < 16; ++t) {
        int k = kbase + t + (hi << 4);
        float f = 0.f;
        if (k < K_real) {
          if (IS_CONV) {
            int c = k >> 4, rq = k & 15;
            f = xb[(size_t)c * H * Wd + (rq >> 2) * Wd + (rq & 3)];
          } else {
            f = xb[k];
          }
        }
        bb[t] = (_Float16)f;
      }
    }
    acc = __builtin_amdgcn_wmma_f32_16x16x32_f16(false, a, false, bb, (short)0, acc, false, false);
  }
  // D layout: lane holds col n=lo, rows mm = v + 8*hi
#pragma unroll
  for (int v = 0; v < 8; ++v) {
    int mm = mt * 16 + v + (hi << 3);
    if (mm < M_real) {
      float val = acc[v] + bias[mm];
      if (relu) val = fmaxf(val, 0.f);
      Y[(size_t)bimg * M_real * OHW + (size_t)mm * OHW + sp] = val;
    }
  }
}

// ---------------------------------------------------------------------------
// Generic stride-1 maxpool (input already ReLU'd by GEMM store).
// ---------------------------------------------------------------------------
__global__ void maxpool_kernel(const float* __restrict__ in, float* __restrict__ out,
                               int C, int IH, int IW, int OH, int OW, int k, int total) {
  int idx = blockIdx.x * blockDim.x + threadIdx.x;
  if (idx >= total) return;
  int ox = idx % OW;
  int t = idx / OW;
  int oy = t % OH; t /= OH;
  int c = t % C;
  int b = t / C;
  const float* ib = in + ((size_t)(b * C + c) * IH + oy) * IW + ox;
  float m = -3.4e38f;
  for (int r = 0; r < k; ++r)
    for (int q = 0; q < k; ++q)
      m = fmaxf(m, ib[r * IW + q]);
  out[idx] = m;
}

// ---------------------------------------------------------------------------
// log_softmax over 2 classes, one thread per batch row.
// ---------------------------------------------------------------------------
__global__ void logsoftmax_kernel(const float* __restrict__ z, float* __restrict__ out) {
  int n = threadIdx.x;  // 256
  float z0 = z[n * 2], z1 = z[n * 2 + 1];
  float m = fmaxf(z0, z1);
  float lse = m + logf(expf(z0 - m) + expf(z1 - m));
  out[n * 2] = z0 - lse;
  out[n * 2 + 1] = z1 - lse;
}

// ---------------------------------------------------------------------------
extern "C" void kernel_launch(void* const* d_in, const int* in_sizes, int n_in,
                              void* d_out, int out_size, void* d_ws, size_t ws_size,
                              hipStream_t stream) {
  const float* x    = (const float*)d_in[0];
  const float* c1w  = (const float*)d_in[1];
  const float* c1b  = (const float*)d_in[2];
  const float* qw   = (const float*)d_in[3];
  const float* c2w  = (const float*)d_in[4];
  const float* c2b  = (const float*)d_in[5];
  const float* c3w  = (const float*)d_in[6];
  const float* c3b  = (const float*)d_in[7];
  const float* c4w  = (const float*)d_in[8];
  const float* c4b  = (const float*)d_in[9];
  const float* f1w  = (const float*)d_in[10];
  const float* f1b  = (const float*)d_in[11];
  const float* f2w  = (const float*)d_in[12];
  const float* f2b  = (const float*)d_in[13];
  const float* f3w  = (const float*)d_in[14];
  const float* f3b  = (const float*)d_in[15];
  const float* f4w  = (const float*)d_in[16];
  const float* f4b  = (const float*)d_in[17];
  float* out = (float*)d_out;

  // workspace carve (256B aligned)
  char* p = (char*)d_ws;
  auto carve = [&](size_t bytes) -> void* {
    void* r = (void*)p;
    p += (bytes + 255) & ~(size_t)255;
    return r;
  };
  float*     c1  = (float*)carve((size_t)256 * 90 * 90 * 4);   // conv1+tanh
  float*     h0  = (float*)carve((size_t)256 * 44 * 44 * 4);   // pooled*pi/2
  _Float16*  Vre = (_Float16*)carve(16 * 32 * 2);              // circuit A-frag (re)
  _Float16*  Vim = (_Float16*)carve(16 * 32 * 2);              // circuit A-frag (im)
  float*     qo  = (float*)carve((size_t)256 * 4 * 22 * 22 * 4);   // quanv+relu
  float*     y2  = (float*)carve((size_t)256 * 32 * 19 * 19 * 4);  // conv2+relu
  float*     p2  = (float*)carve((size_t)256 * 32 * 16 * 16 * 4);  // pool 4s1
  float*     y3  = (float*)carve((size_t)256 * 64 * 13 * 13 * 4);  // conv3+relu
  float*     p3  = (float*)carve((size_t)256 * 64 * 12 * 12 * 4);  // pool 2s1
  float*     y4  = (float*)carve((size_t)256 * 128 * 25 * 4);      // conv4+relu
  float*     h1  = (float*)carve((size_t)256 * 128 * 4);
  float*     h2  = (float*)carve((size_t)256 * 64 * 4);
  float*     h3  = (float*)carve((size_t)256 * 16 * 4);
  float*     lg  = (float*)carve((size_t)256 * 2 * 4);

  conv1_tanh_kernel<<<(256 * 90 * 90 + 255) / 256, 256, 0, stream>>>(x, c1w, c1b, c1);
  pool1_scale_kernel<<<(256 * 44 * 44 + 255) / 256, 256, 0, stream>>>(c1, h0);
  unitary_kernel<<<1, 256, 0, stream>>>(qw, Vre, Vim);
  quanv_kernel<<<968, 256, 0, stream>>>(h0, Vre, Vim, qo);  // 7744 tiles / 8 waves

  // conv2: M=32 K=64  N=256*19*19=92416  -> tiles 2*5776
  gemm_wmma_kernel<true><<<(11552 + 7) / 8, 256, 0, stream>>>(
      c2w, qo, c2b, y2, 32, 64, 2, 11552, 1, 4, 22, 22, 19, 19, 1);
  maxpool_kernel<<<(2097152 + 255) / 256, 256, 0, stream>>>(y2, p2, 32, 19, 19, 16, 16, 4, 2097152);

  // conv3: M=64 K=512 N=256*13*13=43264 -> tiles 4*2704
  gemm_wmma_kernel<true><<<(10816 + 7) / 8, 256, 0, stream>>>(
      c3w, p2, c3b, y3, 64, 512, 4, 10816, 1, 32, 16, 16, 13, 13, 1);
  maxpool_kernel<<<(2359296 + 255) / 256, 256, 0, stream>>>(y3, p3, 64, 13, 13, 12, 12, 2, 2359296);

  // conv4: M=128 K=1024 N=256*25=6400 (stride 2) -> tiles 8*400
  gemm_wmma_kernel<true><<<(3200 + 7) / 8, 256, 0, stream>>>(
      c4w, p3, c4b, y4, 128, 1024, 8, 3200, 1, 64, 12, 12, 5, 5, 2);

  // fc layers (N=256 batch); y4 layout (B,128,5,5) flat == fc1 input [256][3200]
  gemm_wmma_kernel<false><<<(128 + 7) / 8, 256, 0, stream>>>(
      f1w, y4, f1b, h1, 128, 3200, 8, 128, 1, 0, 0, 0, 1, 1, 0);
  gemm_wmma_kernel<false><<<(64 + 7) / 8, 256, 0, stream>>>(
      f2w, h1, f2b, h2, 64, 128, 4, 64, 1, 0, 0, 0, 1, 1, 0);
  gemm_wmma_kernel<false><<<(16 + 7) / 8, 256, 0, stream>>>(
      f3w, h2, f3b, h3, 16, 64, 1, 16, 1, 0, 0, 0, 1, 1, 0);
  gemm_wmma_kernel<false><<<(16 + 7) / 8, 256, 0, stream>>>(
      f4w, h3, f4b, lg, 2, 16, 1, 16, 0, 0, 0, 0, 1, 1, 0);

  logsoftmax_kernel<<<1, 256, 0, stream>>>(lg, out);

  (void)in_sizes; (void)n_in; (void)out_size; (void)ws_size;
}
